// Attention_21818433863773
// MI455X (gfx1250) — matrix operand order repaired and checked by reading, not compiled, not run on hardware
//
#include <hip/hip_runtime.h>
#include <hip/hip_bf16.h>

// ---------------------------------------------------------------------------
// Multi-head attention forward on gfx1250 (CDNA5, wave32, WMMA).
//   qkv = x @ w_qkv + b_qkv ; split heads ; S = QK^T * 1/8 ; softmax ;
//   WA = S @ V ; out = WA @ w_proj + b_proj.  d_out = [out | weighted_avg].
// GEMMs: v_wmma_f32_16x16x32_f16, B tiles double-buffered through LDS via
// GLOBAL_LOAD_ASYNC_TO_LDS_B128 (ASYNCcnt). Flash-attention online softmax.
// ---------------------------------------------------------------------------

typedef __attribute__((ext_vector_type(16))) _Float16 v16h;
typedef __attribute__((ext_vector_type(8)))  _Float16 v8h;
typedef __attribute__((ext_vector_type(8)))  float    v8f;
typedef int v4i __attribute__((vector_size(16)));   // matches builtin proto

#define B_    8
#define P_    1024
#define D_    768
#define H_    12
#define HD_   64
#define ND3_  2304
#define MTOT_ 8192              // B_*P_
#define BHP_  65536             // P_*HD_ per (b,h) slab
#define KSTEPS_ (D_ / 32)       // 24 k-steps of 32

// ---- gfx1250 async copy plumbing ------------------------------------------
#if defined(__gfx1250__) && __has_builtin(__builtin_amdgcn_global_load_async_to_lds_b128)
#define ASYNC_COPY 1
typedef __attribute__((address_space(1))) v4i* gbl_v4i;
typedef __attribute__((address_space(3))) v4i* lds_v4i;
__device__ __forceinline__ void async_b128(const _Float16* g, _Float16* l) {
  // signature (from clang diagnostic): (v4i __device__*, lds dst, int, int)
  __builtin_amdgcn_global_load_async_to_lds_b128((gbl_v4i)g, (lds_v4i)l, 0, 0);
}
#endif

template <int N>
__device__ __forceinline__ void wait_async() {
#if __has_builtin(__builtin_amdgcn_s_wait_asynccnt)
  __builtin_amdgcn_s_wait_asynccnt(N);
#else
  asm volatile("s_wait_asynccnt %0" :: "i"(N) : "memory");
#endif
}

// ---- fragment loaders (contiguous k order, shared by A and B) ---------------
// A-matrix 16x32: lane -> row = lane%16, half = lane/16.
//   elems 0..15 = K = half*16 + 0..15 (contiguous, matches B fragments)
__device__ __forceinline__ v16h load_afrag(const _Float16* __restrict__ base,
                                           int ld, int row, int kb, int half) {
  const _Float16* p = base + (size_t)row * ld + kb + half * 16;
  v8h lo = *(const v8h*)p;
  v8h hi = *(const v8h*)(p + 8);
  v16h r;
#pragma unroll
  for (int i = 0; i < 8; ++i) { r[i] = lo[i]; r[i + 8] = hi[i]; }
  return r;
}

// B-matrix 32x16: lane -> col = lane%16, half = lane/16.
//   elems 0..15 = K = half*16 + 0..15 (16 contiguous halves in memory/LDS)
__device__ __forceinline__ v16h load_bfrag16(const _Float16* p) {
  v8h lo = *(const v8h*)p;
  v8h hi = *(const v8h*)(p + 8);
  v16h r;
#pragma unroll
  for (int i = 0; i < 8; ++i) { r[i] = lo[i]; r[i + 8] = hi[i]; }
  return r;
}

__device__ __forceinline__ v8f wmma_f16(v16h a, v16h b, v8f c) {
  return __builtin_amdgcn_wmma_f32_16x16x32_f16(false, a, false, b,
                                                (short)0, c, false, false);
}

// ---- precision conversion kernels -----------------------------------------
__global__ void conv_f16(const float* __restrict__ in,
                         _Float16* __restrict__ out, int n) {
  int i = blockIdx.x * blockDim.x + threadIdx.x;
  if (i < n) out[i] = (_Float16)in[i];
}

// in: [K, N] f32 row-major  ->  out: [N, K] f16 row-major (transposed)
__global__ void conv_f16_t(const float* __restrict__ in,
                           _Float16* __restrict__ out, int K, int N) {
  int i = blockIdx.x * blockDim.x + threadIdx.x;
  if (i < K * N) {
    int k = i / N, n = i - k * N;
    out[(size_t)n * K + k] = (_Float16)in[i];
  }
}

// ---- shared 64x64 GEMM core with LDS-staged, double-buffered B tiles ------
// Block = 128 threads (4 waves). Wave w computes rows [mtile+w*16, +16).
// B tile per k-step: 64 N-rows x 32 K halves = 4 KB; two buffers in LDS.
// Each thread issues 2 async b128 copies per stage (256 chunks / 128 thr).
__device__ __forceinline__ void gemm_core_64x64(
    const _Float16* __restrict__ A, const _Float16* __restrict__ Wt,
    int mtile, int ntile, int tid, _Float16* ldsb, v8f acc[4]) {
  const int lane = tid & 31;
  const int half = lane >> 4;
  const int l16  = lane & 15;

  auto issue_stage = [&](int buf, int kb) {
#pragma unroll
    for (int c = tid; c < 256; c += 128) {
      int row = c >> 2, part = c & 3;                 // 64 rows x 4 16B chunks
      const _Float16* g = Wt + (size_t)(ntile + row) * D_ + kb + part * 8;
      _Float16* l = ldsb + buf * 2048 + row * 32 + part * 8;
#if defined(ASYNC_COPY)
      async_b128(g, l);
#else
      *(v8h*)l = *(const v8h*)g;                      // load + ds_store_b128
#endif
    }
  };

  issue_stage(0, 0);
  v16h a_cur = load_afrag(A, D_, mtile + l16, 0, half);

  for (int kbi = 0; kbi < KSTEPS_; ++kbi) {
    const int kb  = kbi * 32;
    const int buf = kbi & 1;
    if (kbi + 1 < KSTEPS_) {
      issue_stage(buf ^ 1, kb + 32);                  // prefetch next stage
#if defined(ASYNC_COPY)
      wait_async<2>();                                // current stage landed
#endif
    } else {
#if defined(ASYNC_COPY)
      wait_async<0>();
#endif
    }
    __syncthreads();                                  // tile visible to block

    v16h a_next;
    if (kbi + 1 < KSTEPS_) {
      a_next = load_afrag(A, D_, mtile + l16, kb + 32, half);
      if (kbi + 2 < KSTEPS_)
        __builtin_prefetch(A + (size_t)(mtile + l16) * D_ + kb + 64, 0, 0);
    }
#pragma unroll
    for (int j = 0; j < 4; ++j) {
      v16h b = load_bfrag16(ldsb + buf * 2048 + (j * 16 + l16) * 32 + half * 16);
      acc[j] = wmma_f16(a_cur, b, acc[j]);
    }
    a_cur = a_next;
    __syncthreads();                                  // done reading buf
  }
}

// ---- kernel 1: QKV projection GEMM ----------------------------------------
// C[8192, 2304] = xh @ wqkv^T (+bias), scattered into per-head f16 buffers:
//   q,k : [b*H+h][p][64]   vT : [b*H+h][64][p].  Grid (128, 36) x 128 thr.
__global__ __launch_bounds__(128)
void qkv_gemm(const _Float16* __restrict__ xh,     // [8192, 768]
              const _Float16* __restrict__ wt,     // [2304, 768] (transposed)
              const float*    __restrict__ bias,   // [2304]
              _Float16* __restrict__ qbuf,
              _Float16* __restrict__ kbuf,
              _Float16* __restrict__ vtbuf) {
  __shared__ _Float16 ldsb[2 * 2048];
  const int tid  = threadIdx.x;
  const int lane = tid & 31, wave = tid >> 5;
  const int half = lane >> 4, l16 = lane & 15;
  const int mtile = blockIdx.x * 64 + wave * 16;
  const int ntile = blockIdx.y * 64;

  v8f acc[4] = {{}, {}, {}, {}};
  gemm_core_64x64(xh, wt, mtile, ntile, tid, ldsb, acc);

#pragma unroll
  for (int j = 0; j < 4; ++j) {
    int n = ntile + j * 16 + l16;
    float bn = bias[n];
    int which = n / D_;
    int d = n - which * D_;
    int h = d >> 6, hd = d & 63;
#pragma unroll
    for (int r = 0; r < 8; ++r) {
      int m = mtile + r + half * 8;
      int bb = m >> 10, p = m & 1023;
      size_t bh = (size_t)(bb * H_ + h);
      _Float16 val = (_Float16)(acc[j][r] + bn);
      if (which == 0)      qbuf[bh * BHP_ + (size_t)p * HD_ + hd] = val;
      else if (which == 1) kbuf[bh * BHP_ + (size_t)p * HD_ + hd] = val;
      else                 vtbuf[bh * BHP_ + (size_t)hd * P_ + p] = val;
    }
  }
}

// ---- kernel 2: flash attention --------------------------------------------
// One wave per (b, h, 16-query tile). Grid = 8*12*64 = 6144 blocks x 32 thr.
// Online softmax over 32-key blocks; P x P never materialized.
__global__ __launch_bounds__(32)
void flash_attn(const _Float16* __restrict__ qbuf,
                const _Float16* __restrict__ kbuf,
                const _Float16* __restrict__ vtbuf,
                float*    __restrict__ wa_out,     // d_out + 8192*768
                _Float16* __restrict__ wa_h) {     // [8192, 768] f16 scratch
  __shared__ _Float16 lds_p[16 * 32];              // P tile restripe buffer

  const int lane = threadIdx.x & 31;
  const int half = lane >> 4;
  const int l16  = lane & 15;
  const int bid  = blockIdx.x;
  const int qt   = bid & 63;
  const int h    = (bid >> 6) % H_;
  const int b    = bid / (64 * H_);
  const size_t bh = (size_t)(b * H_ + h);

  const _Float16* Q  = qbuf  + bh * BHP_;
  const _Float16* Kc = kbuf  + bh * BHP_;
  const _Float16* VT = vtbuf + bh * BHP_;

  const int qrow = qt * 16 + l16;
  v16h qa0 = load_afrag(Q, HD_, qrow, 0,  half);   // d = 0..31
  v16h qa1 = load_afrag(Q, HD_, qrow, 32, half);   // d = 32..63

  float mi[8], li[8];
  v8f o[4] = {{}, {}, {}, {}};
#pragma unroll
  for (int r = 0; r < 8; ++r) { mi[r] = -1e30f; li[r] = 0.0f; }

  const float scale = 0.125f;                      // HD^-0.5

  for (int kt = 0; kt < P_ / 32; ++kt) {
    // S[16 x 32] as two 16x16 C tiles (keys kt*32+0..15 and +16..31)
    v8f s0 = {}, s1 = {};
    {
      v16h bb = load_bfrag16(Kc + (size_t)(kt * 32 + l16) * HD_ + half * 16);
      s0 = wmma_f16(qa0, bb, s0);
      bb = load_bfrag16(Kc + (size_t)(kt * 32 + l16) * HD_ + 32 + half * 16);
      s0 = wmma_f16(qa1, bb, s0);
    }
    {
      v16h bb = load_bfrag16(Kc + (size_t)(kt * 32 + 16 + l16) * HD_ + half * 16);
      s1 = wmma_f16(qa0, bb, s1);
      bb = load_bfrag16(Kc + (size_t)(kt * 32 + 16 + l16) * HD_ + 32 + half * 16);
      s1 = wmma_f16(qa1, bb, s1);
    }

    // online softmax: row M = r + 8*half lives across the 16 lanes of a half
    float p0[8], p1[8];
#pragma unroll
    for (int r = 0; r < 8; ++r) {
      float a0 = s0[r] * scale, a1 = s1[r] * scale;
      float mx = fmaxf(a0, a1);
#pragma unroll
      for (int sh = 8; sh >= 1; sh >>= 1) mx = fmaxf(mx, __shfl_xor(mx, sh, 32));
      float nm   = fmaxf(mi[r], mx);
      float corr = __expf(mi[r] - nm);
      float e0 = __expf(a0 - nm), e1 = __expf(a1 - nm);
      float rs = e0 + e1;
#pragma unroll
      for (int sh = 8; sh >= 1; sh >>= 1) rs += __shfl_xor(rs, sh, 32);
      li[r] = li[r] * corr + rs;
      mi[r] = nm;
      p0[r] = e0; p1[r] = e1;
#pragma unroll
      for (int j = 0; j < 4; ++j) o[j][r] *= corr;
    }

    // restripe P: C layout -> A-fragment layout via LDS (1 wave, 1 KB)
#pragma unroll
    for (int r = 0; r < 8; ++r) {
      int row = r + half * 8;
      lds_p[row * 32 + l16]      = (_Float16)p0[r];
      lds_p[row * 32 + 16 + l16] = (_Float16)p1[r];
    }
    __syncthreads();
    v16h pa;
    {
      v8h lo = *(const v8h*)&lds_p[l16 * 32 + half * 16];
      v8h hi = *(const v8h*)&lds_p[l16 * 32 + half * 16 + 8];
#pragma unroll
      for (int i = 0; i < 8; ++i) { pa[i] = lo[i]; pa[i + 8] = hi[i]; }
    }
    __syncthreads();

    // O[16 x 64] += P[16 x 32] @ V[32 x 64]  (V^T makes K contiguous)
#pragma unroll
    for (int j = 0; j < 4; ++j) {
      v16h vb = load_bfrag16(VT + (size_t)(j * 16 + l16) * P_
                                + kt * 32 + half * 16);
      o[j] = wmma_f16(pa, vb, o[j]);
    }
  }

  // normalize + store weighted_avg (f32 to d_out, f16 for proj GEMM)
#pragma unroll
  for (int r = 0; r < 8; ++r) {
    float inv = 1.0f / li[r];
    int prow = qt * 16 + r + half * 8;
    size_t rowbase = ((size_t)b * P_ + prow) * D_ + h * HD_;
#pragma unroll
    for (int j = 0; j < 4; ++j) {
      float val = o[j][r] * inv;
      wa_out[rowbase + j * 16 + l16] = val;
      wa_h[rowbase + j * 16 + l16]   = (_Float16)val;
    }
  }
}

// ---- kernel 3: output projection GEMM -------------------------------------
// out[8192, 768] = wa_h @ wproj^T + b_proj.  Grid (128, 12), 128 threads.
__global__ __launch_bounds__(128)
void proj_gemm(const _Float16* __restrict__ ah,    // [8192, 768]
               const _Float16* __restrict__ wt,    // [768, 768] (transposed)
               const float*    __restrict__ bias,  // [768]
               float* __restrict__ out) {
  __shared__ _Float16 ldsb[2 * 2048];
  const int tid  = threadIdx.x;
  const int lane = tid & 31, wave = tid >> 5;
  const int half = lane >> 4, l16 = lane & 15;
  const int mtile = blockIdx.x * 64 + wave * 16;
  const int ntile = blockIdx.y * 64;

  v8f acc[4] = {{}, {}, {}, {}};
  gemm_core_64x64(ah, wt, mtile, ntile, tid, ldsb, acc);

#pragma unroll
  for (int j = 0; j < 4; ++j) {
    int n = ntile + j * 16 + l16;
    float bn = bias[n];
#pragma unroll
    for (int r = 0; r < 8; ++r) {
      int m = mtile + r + half * 8;
      out[(size_t)m * D_ + n] = acc[j][r] + bn;
    }
  }
}

// ---------------------------------------------------------------------------
extern "C" void kernel_launch(void* const* d_in, const int* in_sizes, int n_in,
                              void* d_out, int out_size, void* d_ws,
                              size_t ws_size, hipStream_t stream) {
  (void)in_sizes; (void)n_in; (void)out_size; (void)ws_size;
  const float* x      = (const float*)d_in[0];
  const float* w_qkv  = (const float*)d_in[1];
  const float* b_qkv  = (const float*)d_in[2];
  const float* w_proj = (const float*)d_in[3];
  const float* b_proj = (const float*)d_in[4];

  float* out    = (float*)d_out;                       // [8192, 768]
  float* wa_out = out + (size_t)MTOT_ * D_;            // [8192, 768]

  // f16 workspace carve-up (~68 MB)
  _Float16* ws     = (_Float16*)d_ws;
  _Float16* xh     = ws;                               // 6291456
  _Float16* wqkvt  = xh + (size_t)MTOT_ * D_;          // 1769472
  _Float16* wprojt = wqkvt + (size_t)ND3_ * D_;        //  589824
  _Float16* qb     = wprojt + (size_t)D_ * D_;         // 6291456
  _Float16* kb     = qb + (size_t)B_ * H_ * P_ * HD_;
  _Float16* vt     = kb + (size_t)B_ * H_ * P_ * HD_;
  _Float16* wah    = vt + (size_t)B_ * H_ * P_ * HD_;

  const int nx = MTOT_ * D_;
  conv_f16  <<<(nx + 255) / 256, 256, 0, stream>>>(x, xh, nx);
  conv_f16_t<<<(D_ * ND3_ + 255) / 256, 256, 0, stream>>>(w_qkv, wqkvt, D_, ND3_);
  conv_f16_t<<<(D_ * D_  + 255) / 256, 256, 0, stream>>>(w_proj, wprojt, D_, D_);

  dim3 g1(MTOT_ / 64, ND3_ / 64);                      // (128, 36)
  qkv_gemm<<<g1, 128, 0, stream>>>(xh, wqkvt, b_qkv, qb, kb, vt);

  flash_attn<<<B_ * H_ * (P_ / 16), 32, 0, stream>>>(qb, kb, vt, wa_out, wah);

  dim3 g3(MTOT_ / 64, D_ / 64);                        // (128, 12)
  proj_gemm<<<g3, 128, 0, stream>>>(wah, wprojt, b_proj, out);
}
